// SelfAttention_4578435138065
// MI455X (gfx1250) — matrix operand order repaired and checked by reading, not compiled, not run on hardware
//
#include <hip/hip_runtime.h>
#include <hip/hip_bf16.h>

// ---------------------------------------------------------------------------
// Self-attention (B=4, C=192, Ci=24, H=W=64, L=4096), flash-attention style.
//   Stage 1: Q/K/V 1x1-conv projections, f32 -> bf16, laid out for WMMA.
//   Stage 2: per-16-query-tile online-softmax attention, transposed mapping:
//            S^T = K·Q^T and O^T = V^T·P^T so softmax runs within lanes.
//            64-key chunks: 8 x v_wmma_f32_16x16x32_bf16 per iteration.
//   Stage 3: att = sigmoid(Wout @ O); y = att*x + x.
// ---------------------------------------------------------------------------

#define BATCH 4
#define CH    192
#define CI    24
#define CIP   32          // Ci padded to WMMA K
#define LL    4096        // H*W

typedef __attribute__((ext_vector_type(16))) __bf16        v16bf;
typedef __attribute__((ext_vector_type(8)))  float         v8f;
typedef __attribute__((ext_vector_type(4)))  float         f32x4;
typedef __attribute__((ext_vector_type(4)))  unsigned int  u32x4;

union BfOp {
  v16bf v;
  u32x4 u[2];
  unsigned int d[8];
};

static __device__ __forceinline__ unsigned short f2bf(float f) {
  unsigned int u = __builtin_bit_cast(unsigned int, f);
  u += 0x7FFFu + ((u >> 16) & 1u);          // round-to-nearest-even
  return (unsigned short)(u >> 16);
}

// pack two f32 -> packed bf16 dword: low16 = a, high16 = b  (RNE)
static __device__ __forceinline__ unsigned int pack2bf(float a, float b) {
  unsigned int ua = __builtin_bit_cast(unsigned int, a);
  unsigned int ub = __builtin_bit_cast(unsigned int, b);
  ua += 0x7FFFu + ((ua >> 16) & 1u);
  ub += 0x7FFFu + ((ub >> 16) & 1u);
  return (ua >> 16) | (ub & 0xFFFF0000u);
}

// lane <-> lane^16 exchange (SWAPX16), DS swizzle, no LDS memory touched
static __device__ __forceinline__ unsigned int swz16u(unsigned int x) {
  return (unsigned int)__builtin_amdgcn_ds_swizzle((int)x, 0x401F);
}
static __device__ __forceinline__ float swz16f(float x) {
  return __builtin_bit_cast(float,
      __builtin_amdgcn_ds_swizzle(__builtin_bit_cast(int, x), 0x401F));
}

static __device__ __forceinline__ v8f wmma_bf16(const BfOp& a, const BfOp& b,
                                                v8f c) {
  return __builtin_amdgcn_wmma_f32_16x16x32_bf16(false, a.v, false, b.v,
                                                 (short)0, c, false, false);
}

// ---------------------------------------------------------------------------
// Stage 1: projections. One thread per pixel; weights (transposed) in LDS.
// Outputs:
//   Qbf, Kbf : bf16 [B][L][32]  (row = pixel, channels contiguous, 24..31 = 0)
//   Vbf      : bf16 [B][32][L]  (channel-major, rows 24..31 = 0)
// ---------------------------------------------------------------------------
__global__ __launch_bounds__(256) void proj_kernel(
    const float* __restrict__ x,  const float* __restrict__ Wq,
    const float* __restrict__ Wk, const float* __restrict__ Wv,
    unsigned short* __restrict__ Qbf, unsigned short* __restrict__ Kbf,
    unsigned short* __restrict__ Vbf)
{
  __shared__ float wq[CH * CI], wk[CH * CI], wv[CH * CI];
  const int t = threadIdx.x;
  for (int i = t; i < CH * CI; i += 256) {
    const int ci = i % CI, c = i / CI;      // lds layout [c][ci]
    wq[i] = Wq[ci * CH + c];
    wk[i] = Wk[ci * CH + c];
    wv[i] = Wv[ci * CH + c];
  }
  __syncthreads();

  const int p = blockIdx.x * 256 + t;       // global pixel (block never spans b)
  const int b = p >> 12;
  const int l = p & (LL - 1);
  const float* xb = x + (size_t)b * CH * LL + l;

  float aq[CI], ak[CI], av[CI];
#pragma unroll
  for (int j = 0; j < CI; ++j) { aq[j] = 0.f; ak[j] = 0.f; av[j] = 0.f; }

  for (int c = 0; c < CH; ++c) {
    const float xv = xb[(size_t)c * LL];    // coalesced across the block
    const float* wqc = &wq[c * CI];
    const float* wkc = &wk[c * CI];
    const float* wvc = &wv[c * CI];
#pragma unroll
    for (int j = 0; j < CI; ++j) {
      aq[j] = fmaf(wqc[j], xv, aq[j]);
      ak[j] = fmaf(wkc[j], xv, ak[j]);
      av[j] = fmaf(wvc[j], xv, av[j]);
    }
  }

  // Q/K: packed dword stores, rows of 64 B
  unsigned int* qrow = (unsigned int*)(Qbf + (size_t)p * CIP);
  unsigned int* krow = (unsigned int*)(Kbf + (size_t)p * CIP);
#pragma unroll
  for (int j = 0; j < CI / 2; ++j) {
    qrow[j] = pack2bf(aq[2 * j], aq[2 * j + 1]);
    krow[j] = pack2bf(ak[2 * j], ak[2 * j + 1]);
  }
#pragma unroll
  for (int j = CI / 2; j < CIP / 2; ++j) { qrow[j] = 0u; krow[j] = 0u; }

  // V: channel-major, strided b16 stores (pad rows zeroed)
  unsigned short* vb = Vbf + (size_t)b * CIP * LL + l;
#pragma unroll
  for (int j = 0; j < CI; ++j) vb[(size_t)j * LL] = f2bf(av[j]);
#pragma unroll
  for (int j = CI; j < CIP; ++j) vb[(size_t)j * LL] = 0;
}

// ---------------------------------------------------------------------------
// Stage 2: flash attention, transposed tiles. 8 waves/block, one 16-query
// tile per wave. Per 64-key chunk: 4 score WMMAs (S^T) + in-lane softmax +
// 4 PV WMMAs (O^T). Softmax state is one scalar per lane (its query column).
// Omid: f32 [B][L][32] (cols 0..23 valid).
// ---------------------------------------------------------------------------
__global__ __launch_bounds__(256) void attn_kernel(
    const unsigned short* __restrict__ Qbf,
    const unsigned short* __restrict__ Kbf,
    const unsigned short* __restrict__ Vbf,
    float* __restrict__ Omid)
{
  const int lane = threadIdx.x & 31;
  const int wave = threadIdx.x >> 5;
  const int T    = blockIdx.x * 8 + wave;     // global query-tile id
  const int b    = T >> 8;                    // 256 tiles per batch
  const int qb   = (T & 255) << 4;            // first query of tile
  const int half = lane >> 4;                 // 0 | 1
  const int l15  = lane & 15;

  // B operand: Q^T (32 ch x 16 queries). Lane n = query column qb+n,
  // lanes 0-15 hold ch 0-7,16-23, lanes 16-31 hold ch 8-15,24-31 (ISA layout).
  BfOp bq;
  {
    const unsigned short* qr = Qbf + (size_t)(b * LL + qb + l15) * CIP + half * 8;
    bq.u[0] = *(const u32x4*)(qr);
    bq.u[1] = *(const u32x4*)(qr + 16);
  }

  float m = -3.0e38f, sden = 0.f;             // per-query (per-lane) state
  v8f o0 = {}, o1 = {};                       // O^T tiles: ci 0-15, 16-31

  const unsigned short* Kb = Kbf + (size_t)b * LL * CIP;
  const unsigned short* Vb = Vbf + (size_t)b * CIP * LL;

  for (int kc = 0; kc < LL; kc += 64) {
    // A operands: 4 K tiles (16 keys x 32 ch each), ISA 16-bit A layout
    BfOp ak0, ak1, ak2, ak3;
    {
      const unsigned short* kr = Kb + (size_t)(kc + l15) * CIP + half * 8;
      ak0.u[0] = *(const u32x4*)(kr);
      ak0.u[1] = *(const u32x4*)(kr + 16);
      ak1.u[0] = *(const u32x4*)(kr + 16 * CIP);
      ak1.u[1] = *(const u32x4*)(kr + 16 * CIP + 16);
      ak2.u[0] = *(const u32x4*)(kr + 32 * CIP);
      ak2.u[1] = *(const u32x4*)(kr + 32 * CIP + 16);
      ak3.u[0] = *(const u32x4*)(kr + 48 * CIP);
      ak3.u[1] = *(const u32x4*)(kr + 48 * CIP + 16);
      __builtin_prefetch(kr + 64 * CIP, 0, 0);   // next chunk of K
    }
    const v8f z = {};
    // S^T tiles: s{i} = keys kc+16i .. kc+16i+15 (f32), independent WMMAs
    v8f s0 = wmma_bf16(ak0, bq, z);
    v8f s1 = wmma_bf16(ak1, bq, z);
    v8f s2 = wmma_bf16(ak2, bq, z);
    v8f s3 = wmma_bf16(ak3, bq, z);

    // ---- online softmax, per-lane (query column) ----
    float cmax = s0[0];
#pragma unroll
    for (int v = 1; v < 8; ++v) cmax = fmaxf(cmax, s0[v]);
#pragma unroll
    for (int v = 0; v < 8; ++v) cmax = fmaxf(cmax, s1[v]);
#pragma unroll
    for (int v = 0; v < 8; ++v) cmax = fmaxf(cmax, s2[v]);
#pragma unroll
    for (int v = 0; v < 8; ++v) cmax = fmaxf(cmax, s3[v]);
    cmax = fmaxf(cmax, swz16f(cmax));           // combine lane pair
    const float mn    = fmaxf(m, cmax);
    const float alpha = __expf(m - mn);
    m = mn;

    float p0[8], p1[8], p2[8], p3[8], csum = 0.f;
#pragma unroll
    for (int v = 0; v < 8; ++v) {
      p0[v] = __expf(s0[v] - mn);
      p1[v] = __expf(s1[v] - mn);
      p2[v] = __expf(s2[v] - mn);
      p3[v] = __expf(s3[v] - mn);
      csum += (p0[v] + p1[v]) + (p2[v] + p3[v]);
    }
    csum += swz16f(csum);
    sden = sden * alpha + csum;

    // ---- build B operands P^T (32 keys x 16 q) via xor-16 exchanges ----
    unsigned int pk0[4], pk1[4], pk2[4], pk3[4];
#pragma unroll
    for (int j = 0; j < 4; ++j) {
      pk0[j] = pack2bf(p0[2 * j], p0[2 * j + 1]);
      pk1[j] = pack2bf(p1[2 * j], p1[2 * j + 1]);
      pk2[j] = pack2bf(p2[2 * j], p2[2 * j + 1]);
      pk3[j] = pack2bf(p3[2 * j], p3[2 * j + 1]);
    }
    BfOp bpA, bpB;                               // keys kc..+31, kc+32..+63
#pragma unroll
    for (int j = 0; j < 4; ++j) {
      // lanes<16 send their odd tile (keys +16..23); lanes>=16 send even tile
      const unsigned int sendA = half ? pk0[j] : pk1[j];
      const unsigned int recvA = swz16u(sendA);
      bpA.d[j]     = half ? recvA  : pk0[j];     // keys 0-7   | 16-23
      bpA.d[4 + j] = half ? pk1[j] : recvA;      // keys 8-15  | 24-31
      const unsigned int sendB = half ? pk2[j] : pk3[j];
      const unsigned int recvB = swz16u(sendB);
      bpB.d[j]     = half ? recvB  : pk2[j];     // keys 32-39 | 48-55
      bpB.d[4 + j] = half ? pk3[j] : recvB;      // keys 40-47 | 56-63
    }

    // A operands: V^T tiles (16 ci x 32 keys) from channel-major [32][L]
    BfOp av00, av10, av01, av11;                 // [ci-half][key-half]
    {
      const unsigned short* vr = Vb + (size_t)l15 * LL + kc + half * 16;
      av00.u[0] = *(const u32x4*)(vr);
      av00.u[1] = *(const u32x4*)(vr + 8);
      av01.u[0] = *(const u32x4*)(vr + 32);
      av01.u[1] = *(const u32x4*)(vr + 40);
      const unsigned short* vr1 = vr + (size_t)16 * LL;
      av10.u[0] = *(const u32x4*)(vr1);
      av10.u[1] = *(const u32x4*)(vr1 + 8);
      av11.u[0] = *(const u32x4*)(vr1 + 32);
      av11.u[1] = *(const u32x4*)(vr1 + 40);
      __builtin_prefetch(vr + 64, 0, 0);         // next chunk of V
    }

    // rescale accumulators (alpha uniform per lane) and accumulate O^T
#pragma unroll
    for (int v = 0; v < 8; ++v) { o0[v] *= alpha; o1[v] *= alpha; }
    o0 = wmma_bf16(av00, bpA, o0);               // o0/o1 interleaved: each
    o1 = wmma_bf16(av10, bpA, o1);               // WMMA has an independent op
    o0 = wmma_bf16(av01, bpB, o0);               // between RAW accumulations
    o1 = wmma_bf16(av11, bpB, o1);
  }

  // Normalize and write O (f32 [B][L][32]): lane's query column is contiguous
  // in ci. tile0 -> ci half*8..half*8+7 ; tile1 -> ci 16..23 (half==0 only).
  const float inv = 1.0f / sden;
  float* ob = Omid + (size_t)(b * LL + qb + l15) * CIP;
  f32x4 w;
  w = (f32x4){o0[0] * inv, o0[1] * inv, o0[2] * inv, o0[3] * inv};
  *(f32x4*)(ob + half * 8) = w;
  w = (f32x4){o0[4] * inv, o0[5] * inv, o0[6] * inv, o0[7] * inv};
  *(f32x4*)(ob + half * 8 + 4) = w;
  if (half == 0) {
    w = (f32x4){o1[0] * inv, o1[1] * inv, o1[2] * inv, o1[3] * inv};
    *(f32x4*)(ob + 16) = w;
    w = (f32x4){o1[4] * inv, o1[5] * inv, o1[6] * inv, o1[7] * inv};
    *(f32x4*)(ob + 20) = w;
  }
}

// ---------------------------------------------------------------------------
// Stage 3: att = sigmoid(Wout @ O); y = att * x + x. One thread per pixel.
// Wout is [C][Ci] row-major (already the layout we want in LDS).
// ---------------------------------------------------------------------------
__global__ __launch_bounds__(256) void out_kernel(
    const float* __restrict__ x, const float* __restrict__ Wout,
    const float* __restrict__ Omid, float* __restrict__ y)
{
  __shared__ float wl[CH * CI];
  const int t = threadIdx.x;
  for (int i = t; i < CH * CI; i += 256) wl[i] = Wout[i];
  __syncthreads();

  const int p = blockIdx.x * 256 + t;
  const int b = p >> 12;
  const int l = p & (LL - 1);

  float o[CI];
  const float* orow = Omid + (size_t)p * CIP;
#pragma unroll
  for (int j = 0; j < CI; ++j) o[j] = orow[j];

  const float* xb = x + (size_t)b * CH * LL + l;
  float*       yb = y + (size_t)b * CH * LL + l;
  for (int c = 0; c < CH; ++c) {
    const float* wc = &wl[c * CI];
    float acc = 0.f;
#pragma unroll
    for (int j = 0; j < CI; ++j) acc = fmaf(wc[j], o[j], acc);
    const float att = 1.0f / (1.0f + __expf(-acc));
    const float xv  = xb[(size_t)c * LL];
    yb[(size_t)c * LL] = att * xv + xv;
  }
}

// ---------------------------------------------------------------------------
extern "C" void kernel_launch(void* const* d_in, const int* in_sizes, int n_in,
                              void* d_out, int out_size, void* d_ws, size_t ws_size,
                              hipStream_t stream) {
  const float* x    = (const float*)d_in[0];
  const float* Wq   = (const float*)d_in[1];
  const float* Wk   = (const float*)d_in[2];
  const float* Wv   = (const float*)d_in[3];
  const float* Wout = (const float*)d_in[4];
  float* y = (float*)d_out;

  // Workspace carve: 3 x 1MB bf16 (Q,K,V) + 2MB f32 (O) = 5MB.
  unsigned short* Qbf = (unsigned short*)d_ws;
  unsigned short* Kbf = Qbf + (size_t)BATCH * LL * CIP;
  unsigned short* Vbf = Kbf + (size_t)BATCH * LL * CIP;
  float*          Om  = (float*)(Vbf + (size_t)BATCH * LL * CIP);

  const int npix = BATCH * LL;                 // 16384
  proj_kernel<<<npix / 256, 256, 0, stream>>>(x, Wq, Wk, Wv, Qbf, Kbf, Vbf);

  const int ntiles = BATCH * (LL / 16);        // 1024 query tiles, 8 waves/block
  attn_kernel<<<ntiles / 8, 256, 0, stream>>>(Qbf, Kbf, Vbf, Om);

  out_kernel<<<npix / 256, 256, 0, stream>>>(x, Wout, Om, y);
}
